// SingleHeadAttention_7516192768863
// MI455X (gfx1250) — matrix operand order repaired and checked by reading, compiled, not run on hardware
//
#include <hip/hip_runtime.h>

typedef __attribute__((ext_vector_type(16))) __bf16 v16bf;
typedef __attribute__((ext_vector_type(8)))  __bf16 v8bf;
typedef __attribute__((ext_vector_type(8)))  float  v8f;

#define B_  8
#define S_  2048
#define NU  1024
#define DK  64
#define BS  (B_ * S_)

static __device__ __forceinline__ v16bf cat16(v8bf lo, v8bf hi) {
  return __builtin_shufflevector(lo, hi, 0, 1, 2, 3, 4, 5, 6, 7,
                                 8, 9, 10, 11, 12, 13, 14, 15);
}

// ---------------------------------------------------------------------------
// Kernel 0: one-shot weight conversion f32 -> bf16 (3 x 64 x 1024, 384 KB).
// ---------------------------------------------------------------------------
__global__ __launch_bounds__(256)
void wconv_kernel(const float* __restrict__ Wq, const float* __restrict__ Wk,
                  const float* __restrict__ Wv,
                  __bf16* __restrict__ Wqb, __bf16* __restrict__ Wkb,
                  __bf16* __restrict__ Wvb)
{
  const int i = blockIdx.x * 256 + threadIdx.x;          // 0 .. 65535
  const float*  src = (blockIdx.y == 0) ? Wq  : (blockIdx.y == 1) ? Wk  : Wv;
  __bf16*       dst = (blockIdx.y == 0) ? Wqb : (blockIdx.y == 1) ? Wkb : Wvb;
  dst[i] = (__bf16)src[i];
}

// ---------------------------------------------------------------------------
// Kernel 1: fused Q/K/V projection.  One 32-row tile of X per block, one wave
// per projection (q,k,v); each wave computes TWO 16-row A-tiles so every
// loaded weight B-tile feeds 2 WMMAs (halves L2 weight traffic).
// Q is pre-scaled by 1/sqrt(d_k); V is stored transposed (DK x BS) via
// wave-uniform strides (branch-free epilogue).
// Streams the 201 MB of f32 activations exactly once (HBM-bound phase).
// ---------------------------------------------------------------------------
__global__ __launch_bounds__(96)
void qkv_proj_kernel(const float* __restrict__ q_in, const float* __restrict__ k_in,
                     const float* __restrict__ v_in,
                     const __bf16* __restrict__ Wqb, const float* __restrict__ bq,
                     const __bf16* __restrict__ Wkb, const float* __restrict__ bk,
                     const __bf16* __restrict__ Wvb, const float* __restrict__ bv,
                     __bf16* __restrict__ Qo, __bf16* __restrict__ Ko,
                     __bf16* __restrict__ Vt)
{
  const int wave = threadIdx.x >> 5;
  const int lane = threadIdx.x & 31;
  const float*  X    = (wave == 0) ? q_in : (wave == 1) ? k_in : v_in;
  const __bf16* W    = (wave == 0) ? Wqb  : (wave == 1) ? Wkb  : Wvb;
  const float*  bias = (wave == 0) ? bq   : (wave == 1) ? bk   : bv;
  __bf16*       dst  = (wave == 0) ? Qo   : (wave == 1) ? Ko   : Vt;
  // Q/K: row-major (row*DK + col); V: transposed (col*BS + row)
  const size_t  rstride = (wave == 2) ? (size_t)1  : (size_t)DK;
  const size_t  cstride = (wave == 2) ? (size_t)BS : (size_t)1;
  const float   qscale  = (wave == 0) ? 0.125f : 1.0f;

  const int row0 = blockIdx.x * 32;       // flattened over B*S
  const int m    = lane & 15;
  const int half = lane >> 4;
  const int kb   = half * 16;

  v8f acc[2][4] = {};                     // two 16x64 f32 accumulators

  for (int k0 = 0; k0 < NU; k0 += 32) {
    // ---- two A tiles: X[row0+t*16 ..][k0..+31] f32 -> bf16, A-layout ------
    v16bf a[2];
#pragma unroll
    for (int t = 0; t < 2; ++t) {
      const float* arow = X + (size_t)(row0 + t * 16 + m) * NU + k0 + half * 8;
#pragma unroll
      for (int h = 0; h < 8; ++h) a[t][h]     = (__bf16)arow[h];
#pragma unroll
      for (int h = 0; h < 8; ++h) a[t][8 + h] = (__bf16)arow[16 + h];
    }

    // ---- 4 B tiles: bf16 weights, each feeds both A tiles -----------------
#pragma unroll
    for (int nt = 0; nt < 4; ++nt) {
      const v16bf bm =
          cat16(*(const v8bf*)(W + (size_t)(nt * 16 + m) * NU + k0 + half * 8),
                *(const v8bf*)(W + (size_t)(nt * 16 + m) * NU + k0 + 16 + half * 8));
      acc[0][nt] = __builtin_amdgcn_wmma_f32_16x16x32_bf16(
          false, a[0], false, bm, (short)0, acc[0][nt], false, false);
      acc[1][nt] = __builtin_amdgcn_wmma_f32_16x16x32_bf16(
          false, a[1], false, bm, (short)0, acc[1][nt], false, false);
    }
  }

  // epilogue: + bias (Q also * 1/sqrt(dk)), convert to bf16, branch-free store
#pragma unroll
  for (int t = 0; t < 2; ++t) {
#pragma unroll
    for (int nt = 0; nt < 4; ++nt) {
      const int   col  = nt * 16 + m;
      const float bofs = bias[col];
#pragma unroll
      for (int r = 0; r < 8; ++r) {
        const int row = row0 + t * 16 + r + half * 8;
        dst[(size_t)row * rstride + (size_t)col * cstride] =
            (__bf16)((acc[t][nt][r] + bofs) * qscale);
      }
    }
  }
}

// ---------------------------------------------------------------------------
// Kernel 2: causal flash-attention over bf16 Q/K/V^T (6 MB, L2-resident).
// One wave per 16-query tile; keys streamed in chunks of 32 with online
// softmax; P(16x32) goes through LDS to become the next WMMA A-operand.
// ---------------------------------------------------------------------------
__global__ __launch_bounds__(256)
void attention_kernel(const __bf16* __restrict__ Q, const __bf16* __restrict__ K,
                      const __bf16* __restrict__ Vt, float* __restrict__ out)
{
  __shared__ __bf16 sP[8][16 * 32];       // per-wave 16x32 probability tile

  const int wave  = threadIdx.x >> 5;
  const int lane  = threadIdx.x & 31;
  const int tile  = blockIdx.x * 8 + wave;
  const int qrow0 = tile * 16;            // flattened over B*S
  const int batch = qrow0 / S_;
  const int qbase = qrow0 % S_;           // query pos of tile row 0
  const int m     = lane & 15;
  const int half  = lane >> 4;
  const int kb    = half * 16;

  // Q tile in A-layout (already scaled by 1/sqrt(dk) at projection time)
  v16bf qa[2];
  {
    const __bf16* qrow = Q + (size_t)(qrow0 + m) * DK;
#pragma unroll
    for (int s = 0; s < 2; ++s)
      qa[s] = cat16(*(const v8bf*)(qrow + s * 32 + half * 8),
                    *(const v8bf*)(qrow + s * 32 + 16 + half * 8));
  }

  float mi[8], li[8];
#pragma unroll
  for (int r = 0; r < 8; ++r) { mi[r] = -3.0e38f; li[r] = 0.0f; }
  v8f acc[4] = {};

  const __bf16* Kb  = K  + (size_t)batch * S_ * DK;
  const __bf16* Vtb = Vt + (size_t)batch * S_;     // column base within row

  for (int kc = 0; kc <= qbase + 15; kc += 32) {
    // prefetch next key chunk (speculative, gfx1250 global_prefetch_b8)
    __builtin_prefetch(Kb + (size_t)(kc + 32) * DK, 0, 1);

    // ---- scores S = Q * K^T : 2 N-tiles x 2 k-steps -----------------------
    v8f sc[2];
#pragma unroll
    for (int nt = 0; nt < 2; ++nt) {
      v8f z = {};
#pragma unroll
      for (int s = 0; s < 2; ++s) {
        // B column n = key row (kc + nt*16 + m); table order along d
        const v16bf bm = cat16(
            *(const v8bf*)(Kb + (size_t)(kc + nt * 16 + m) * DK + s * 32 + half * 8),
            *(const v8bf*)(Kb + (size_t)(kc + nt * 16 + m) * DK + s * 32 + 16 + half * 8));
        z = __builtin_amdgcn_wmma_f32_16x16x32_bf16(
            false, qa[s], false, bm, (short)0, z, false, false);
      }
      sc[nt] = z;
    }

    // ---- causal mask + online softmax (rowwise over 16-lane groups) -------
#pragma unroll
    for (int r = 0; r < 8; ++r) {
      const int qpos = qbase + r + half * 8;
      const int c0 = kc + m, c1 = kc + 16 + m;
      float s0 = (c0 <= qpos) ? sc[0][r] : -1.0e9f;
      float s1 = (c1 <= qpos) ? sc[1][r] : -1.0e9f;
      float mx = fmaxf(s0, s1);
#pragma unroll
      for (int d = 8; d >= 1; d >>= 1) mx = fmaxf(mx, __shfl_xor(mx, d, 32));
      const float mnew  = fmaxf(mi[r], mx);
      const float scale = __expf(mi[r] - mnew);
      const float p0 = __expf(s0 - mnew);
      const float p1 = __expf(s1 - mnew);
      float rs = p0 + p1;
#pragma unroll
      for (int d = 8; d >= 1; d >>= 1) rs += __shfl_xor(rs, d, 32);
      li[r] = li[r] * scale + rs;
      mi[r] = mnew;
#pragma unroll
      for (int nt = 0; nt < 4; ++nt) acc[nt][r] *= scale;

      const int rowm = r + half * 8;
      sP[wave][rowm * 32 + m]      = (__bf16)p0;
      sP[wave][rowm * 32 + 16 + m] = (__bf16)p1;
    }

    // ---- reload P as A-operand (layout swap via LDS, same wave) -----------
    const v16bf pa = cat16(*(const v8bf*)&sP[wave][m * 32 + half * 8],
                           *(const v8bf*)&sP[wave][m * 32 + 16 + half * 8]);

    // ---- acc += P(16x32) * V(32x64) : 4 N-tiles, contiguous V^T loads -----
#pragma unroll
    for (int nt = 0; nt < 4; ++nt) {
      const v16bf vb =
          cat16(*(const v8bf*)(Vtb + (size_t)(nt * 16 + m) * BS + kc + half * 8),
                *(const v8bf*)(Vtb + (size_t)(nt * 16 + m) * BS + kc + 16 + half * 8));
      acc[nt] = __builtin_amdgcn_wmma_f32_16x16x32_bf16(
          false, pa, false, vb, (short)0, acc[nt], false, false);
    }
  }

  // ---- normalize and store f32 output -------------------------------------
#pragma unroll
  for (int nt = 0; nt < 4; ++nt) {
#pragma unroll
    for (int r = 0; r < 8; ++r) {
      const int M = r + half * 8;
      out[(size_t)(qrow0 + M) * DK + nt * 16 + m] = acc[nt][r] / li[r];
    }
  }
}

// ---------------------------------------------------------------------------
extern "C" void kernel_launch(void* const* d_in, const int* in_sizes, int n_in,
                              void* d_out, int out_size, void* d_ws, size_t ws_size,
                              hipStream_t stream) {
  (void)in_sizes; (void)n_in; (void)out_size; (void)ws_size;

  const float* q_in = (const float*)d_in[0];
  const float* k_in = (const float*)d_in[1];
  const float* v_in = (const float*)d_in[2];
  // d_in[3] is the causal mask; it is implied by indices and never read.
  const float* Wq = (const float*)d_in[4];
  const float* bq = (const float*)d_in[5];
  const float* Wk = (const float*)d_in[6];
  const float* bk = (const float*)d_in[7];
  const float* Wv = (const float*)d_in[8];
  const float* bv = (const float*)d_in[9];

  // workspace layout (bf16):
  //   Qo (BS*DK) | Ko (BS*DK) | Vt (DK*BS) | Wqb | Wkb | Wvb  (~6.4 MB)
  const size_t elems = (size_t)BS * DK;
  const size_t welem = (size_t)DK * NU;
  __bf16* Qo  = (__bf16*)d_ws;
  __bf16* Ko  = Qo + elems;
  __bf16* Vt  = Ko + elems;
  __bf16* Wqb = Vt + elems;
  __bf16* Wkb = Wqb + welem;
  __bf16* Wvb = Wkb + welem;

  wconv_kernel<<<dim3((DK * NU) / 256, 3), 256, 0, stream>>>(
      Wq, Wk, Wv, Wqb, Wkb, Wvb);

  qkv_proj_kernel<<<BS / 32, 96, 0, stream>>>(q_in, k_in, v_in,
                                              Wqb, bq, Wkb, bk, Wvb, bv,
                                              Qo, Ko, Vt);
  attention_kernel<<<(BS / 16) / 8, 256, 0, stream>>>(Qo, Ko, Vt,
                                                      (float*)d_out);
}
